// GraphConvolution_61847529062785
// MI455X (gfx1250) — matrix operand order repaired and checked by reading, not compiled, not run on hardware
//
#include <hip/hip_runtime.h>
#include <hip/hip_bf16.h>

typedef __attribute__((ext_vector_type(16))) _Float16 v16h;
typedef __attribute__((ext_vector_type(8)))  _Float16 v8h;
typedef __attribute__((ext_vector_type(8)))  float    v8f;

#define N_NODES 10000
#define N_EDGES 50000
#define IN_F    1433
#define K_PAD   1440   // next multiple of 32
#define OUT_F   256

// ---------------- 1) agg = noise ----------------
__global__ void k_init(const float* __restrict__ noise, float* __restrict__ agg) {
    size_t i = (size_t)blockIdx.x * blockDim.x + threadIdx.x;
    if (i < (size_t)N_NODES * IN_F) agg[i] = noise[i];
}

// ---------------- 2) scatter-add: one block per edge ----------------
__global__ void k_scatter(const float* __restrict__ feat, const int* __restrict__ edges,
                          float* __restrict__ agg) {
    int e   = blockIdx.x;
    int dst = edges[2 * e + 0];
    int src = edges[2 * e + 1];
    if (src >= N_NODES) return;               // masked invalid edge
    const float* frow = feat + (size_t)src * IN_F;
    float*       arow = agg  + (size_t)dst * IN_F;
    for (int f = threadIdx.x; f < IN_F; f += blockDim.x)
        atomicAdd(&arow[f], frow[f]);
}

// ---------------- 3) per-row L2 normalize -> f16 A (zero-padded K) ----------------
__global__ void k_normalize(const float* __restrict__ agg, _Float16* __restrict__ Ah) {
    int row = blockIdx.x;
    const float* r = agg + (size_t)row * IN_F;
    __shared__ float red[256];
    float s = 0.0f;
    for (int f = threadIdx.x; f < IN_F; f += blockDim.x) { float v = r[f]; s += v * v; }
    red[threadIdx.x] = s;
    __syncthreads();
    for (int st = 128; st > 0; st >>= 1) {
        if ((int)threadIdx.x < st) red[threadIdx.x] += red[threadIdx.x + st];
        __syncthreads();
    }
    float inv = 1.0f / fmaxf(sqrtf(red[0]), 1e-12f);
    _Float16* orow = Ah + (size_t)row * K_PAD;
    for (int f = threadIdx.x; f < K_PAD; f += blockDim.x)
        orow[f] = (f < IN_F) ? (_Float16)(r[f] * inv) : (_Float16)0.0f;
}

// ---------------- 4) weight -> transposed f16 Wt[n][k], zero-padded ----------------
__global__ void k_wconv(const float* __restrict__ W, _Float16* __restrict__ Wt) {
    int idx = blockIdx.x * blockDim.x + threadIdx.x;
    if (idx >= OUT_F * K_PAD) return;
    int n = idx / K_PAD;
    int k = idx % K_PAD;
    Wt[idx] = (k < IN_F) ? (_Float16)W[(size_t)k * OUT_F + n] : (_Float16)0.0f;
}

// ---------------- 5) WMMA GEMM: C = A(10000xK_PAD) * Wt^T + bias ----------------
// One wave computes one 16x16 output tile. 8 waves per block.
__global__ void k_gemm(const _Float16* __restrict__ A, const _Float16* __restrict__ B,
                       const float* __restrict__ bias, float* __restrict__ C) {
    int wave  = threadIdx.x >> 5;                 // 0..7
    int lane  = threadIdx.x & 31;
    int tileN = blockIdx.x * 8 + wave;            // 0..15
    int tileM = blockIdx.y;                       // 0..624
    int l16   = lane & 15;
    int hi    = (lane >= 16) ? 1 : 0;

    // A fragment per ISA layout: lanes 0-15: halves0-7=K+0..7, halves8-15=K+16..23
    //                            lanes16-31: same with K offset +8
    const _Float16* arow = A + (size_t)(tileM * 16 + l16) * K_PAD + (hi ? 8 : 0);
    // B fragment: same K layout as A (lanes 0-15: K+0..7,K+16..23; lanes16-31: +8)
    const _Float16* brow = B + (size_t)(tileN * 16 + l16) * K_PAD + (hi ? 8 : 0);

    v8f c = {};
    for (int k0 = 0; k0 < K_PAD; k0 += 32) {
        v8h a0 = *(const v8h*)(arow + k0);          // K+0..7   (or +8..15 on hi lanes)
        v8h a1 = *(const v8h*)(arow + k0 + 16);     // K+16..23 (or +24..31)
        v8h b0 = *(const v8h*)(brow + k0);          // K+0..7   (or +8..15)
        v8h b1 = *(const v8h*)(brow + k0 + 16);     // K+16..23 (or +24..31)
        v16h av = __builtin_shufflevector(a0, a1, 0,1,2,3,4,5,6,7,8,9,10,11,12,13,14,15);
        v16h bv = __builtin_shufflevector(b0, b1, 0,1,2,3,4,5,6,7,8,9,10,11,12,13,14,15);
        c = __builtin_amdgcn_wmma_f32_16x16x32_f16(
                /*neg_a=*/false, av, /*neg_b=*/false, bv,
                /*c_mod=*/(short)0, c, /*reuse_a=*/false, /*reuse_b=*/false);
    }

    // D layout: VGPR r -> row tileM*16 + r + (hi?8:0), col tileN*16 + l16
    int   n  = tileN * 16 + l16;
    float bn = bias[n];
    int   m0 = tileM * 16 + (hi ? 8 : 0);
    #pragma unroll
    for (int r = 0; r < 8; ++r)
        C[(size_t)(m0 + r) * OUT_F + n] = c[r] + bn;
}

extern "C" void kernel_launch(void* const* d_in, const int* in_sizes, int n_in,
                              void* d_out, int out_size, void* d_ws, size_t ws_size,
                              hipStream_t stream) {
    const float* feat  = (const float*)d_in[0];   // [10000,1433] f32
    const int*   edges = (const int*)  d_in[1];   // [50000,2] int32 (dst,src)
    const float* W     = (const float*)d_in[2];   // [1433,256] f32
    const float* bias  = (const float*)d_in[3];   // [256] f32
    const float* noise = (const float*)d_in[4];   // [10000,1433] f32
    float*       out   = (float*)d_out;           // [10000,256] f32

    char* ws = (char*)d_ws;
    const size_t AH_BYTES = (size_t)N_NODES * K_PAD * sizeof(_Float16);  // 28,800,000
    const size_t WT_BYTES = (size_t)OUT_F   * K_PAD * sizeof(_Float16);  //    737,280
    _Float16* Ah  = (_Float16*)ws;
    _Float16* Wt  = (_Float16*)(ws + AH_BYTES);
    float*    agg = (float*)   (ws + AH_BYTES + WT_BYTES);

    size_t total = (size_t)N_NODES * IN_F;
    k_init<<<dim3((unsigned)((total + 255) / 256)), 256, 0, stream>>>(noise, agg);
    k_scatter<<<N_EDGES, 256, 0, stream>>>(feat, edges, agg);
    k_normalize<<<N_NODES, 256, 0, stream>>>(agg, Ah);
    k_wconv<<<(OUT_F * K_PAD + 255) / 256, 256, 0, stream>>>(W, Wt);
    k_gemm<<<dim3(2, 625), 256, 0, stream>>>(Ah, Wt, bias, out);
}
